// DiversityAttention_63178968924513
// MI455X (gfx1250) — matrix operand order repaired and checked by reading, not compiled, not run on hardware
//
#include <hip/hip_runtime.h>
#include <hip/hip_bf16.h>

// ---------------- problem constants ----------------
#define HID   1024
#define HEADS 16
#define HDIM  64
#define BATCH 2
#define SEQ   2048
#define ROWS  (BATCH * SEQ)        // 4096
#define GAMMA_C 0.5f
#define INV_SQRT_HD 0.125f         // 1/sqrt(64)
#define MASK_BIG 1e30f

typedef __attribute__((ext_vector_type(16))) _Float16 v16h;
typedef __attribute__((ext_vector_type(8)))  _Float16 v8h;
typedef __attribute__((ext_vector_type(8)))  float    v8f;

// ---------------- WMMA helper ----------------
__device__ __forceinline__ v8f wmma_f16(v16h a, v16h b, v8f c) {
    // D = A(16x32 f16) * B(32x16 f16) + C(16x16 f32)
    return __builtin_amdgcn_wmma_f32_16x16x32_f16(
        /*neg_a=*/false, a, /*neg_b=*/false, b,
        /*c_mod=*/(short)0, c, /*reuse_a=*/false, /*reuse_b=*/false);
}

// A-fragment (16x32, MxK) from row-major [rows][rowStride] f16 source.
// lane L (half = L>>4, r = L&15): row = m0 + r
//   halves 0..7  : K = kb + half*8 + h
//   halves 8..15 : K = kb + 16 + half*8 + (h-8)
__device__ __forceinline__ v16h load_a_frag(const _Float16* base, int rowStride,
                                            int m0, int kb) {
    int lane = threadIdx.x & 31;
    int half = lane >> 4;
    int r    = lane & 15;
    const _Float16* p = base + (size_t)(m0 + r) * rowStride + kb + half * 8;
    v8h lo = *(const v8h*)p;
    v8h hi = *(const v8h*)(p + 16);
    v16h out;
#pragma unroll
    for (int i = 0; i < 8; ++i) { out[i] = lo[i]; out[i + 8] = hi[i]; }
    return out;
}

// B-fragment (32x16, KxN) where element B[k][n] = src[(n0+n)*rowStride + kb + k].
// lane L: col = n0 + (L&15); halves 0..7: K = kb + (L>>4)*8 + h; halves 8..15: K = kb + 16 + (L>>4)*8 + (h-8)
__device__ __forceinline__ v16h load_b_frag(const _Float16* base, int rowStride,
                                            int n0, int kb) {
    int lane = threadIdx.x & 31;
    int half = lane >> 4;
    int c    = lane & 15;
    const _Float16* p = base + (size_t)(n0 + c) * rowStride + kb + half * 8;
    v8h lo = *(const v8h*)p;
    v8h hi = *(const v8h*)(p + 16);
    v16h out;
#pragma unroll
    for (int i = 0; i < 8; ++i) { out[i] = lo[i]; out[i + 8] = hi[i]; }
    return out;
}

// ---------------- kernel 1: x -> f16, normalized x -> f16 ----------------
__global__ void k_prep(const float* __restrict__ x,
                       _Float16* __restrict__ xh,
                       _Float16* __restrict__ xnh) {
    int row = blockIdx.x;                      // 0..ROWS-1
    const float* xr = x + (size_t)row * HID;
    __shared__ float red[256];
    float s = 0.f;
    for (int i = threadIdx.x; i < HID; i += 256) { float v = xr[i]; s += v * v; }
    red[threadIdx.x] = s;
    __syncthreads();
    for (int off = 128; off > 0; off >>= 1) {
        if (threadIdx.x < off) red[threadIdx.x] += red[threadIdx.x + off];
        __syncthreads();
    }
    float inv = 1.0f / fmaxf(sqrtf(red[0]), 1e-12f);
    for (int i = threadIdx.x; i < HID; i += 256) {
        float v = xr[i];
        xh [(size_t)row * HID + i] = (_Float16)v;
        xnh[(size_t)row * HID + i] = (_Float16)(v * inv);
    }
}

// ---------------- kernel 2: f32 -> f16 elementwise ----------------
__global__ void k_cvt(const float* __restrict__ in, _Float16* __restrict__ out, int n) {
    int i = blockIdx.x * blockDim.x + threadIdx.x;
    if (i < n) out[i] = (_Float16)in[i];
}

// ---------------- kernel 3: fused QKV GEMM, 64(M)x16(N) per wave x {q,k,v} ----------------
// y = x @ W^T + b ; q,k row-major [m][HID]; v stored transposed [b][h][HDIM][SEQ]
__global__ void __launch_bounds__(256, 2)
k_qkv(const _Float16* __restrict__ xh,
      const _Float16* __restrict__ wq,
      const _Float16* __restrict__ wk,
      const _Float16* __restrict__ wv,
      const float* __restrict__ bq,
      const float* __restrict__ bk,
      const float* __restrict__ bv,
      _Float16* __restrict__ qh,
      _Float16* __restrict__ kh,
      _Float16* __restrict__ vT) {
    int wave = (blockIdx.x * blockDim.x + threadIdx.x) >> 5;   // 0..4095
    int mB = wave >> 6;            // 0..63 (64-row M block)
    int nT = wave & 63;            // 0..63
    int m0 = mB * 64, n0 = nT * 16;

    v8f cq[4] = {{}, {}, {}, {}};
    v8f ck[4] = {{}, {}, {}, {}};
    v8f cv[4] = {{}, {}, {}, {}};
    for (int kb = 0; kb < HID; kb += 32) {
        v16h a[4];
#pragma unroll
        for (int i = 0; i < 4; ++i) a[i] = load_a_frag(xh, HID, m0 + i * 16, kb);
        v16h fq = load_b_frag(wq, HID, n0, kb);
        v16h fk = load_b_frag(wk, HID, n0, kb);
        v16h fv = load_b_frag(wv, HID, n0, kb);
#pragma unroll
        for (int i = 0; i < 4; ++i) {
            cq[i] = wmma_f16(a[i], fq, cq[i]);
            ck[i] = wmma_f16(a[i], fk, ck[i]);
            cv[i] = wmma_f16(a[i], fv, cv[i]);
        }
    }
    int lane = threadIdx.x & 31;
    int colLane = lane & 15, halfSel = lane >> 4;
    int n = n0 + colLane;
    float vbq = bq[n], vbk = bk[n], vbv = bv[n];
    int h = n >> 6, d = n & 63;
#pragma unroll
    for (int i = 0; i < 4; ++i)
#pragma unroll
        for (int r = 0; r < 8; ++r) {
            int m = m0 + i * 16 + r + 8 * halfSel;
            qh[(size_t)m * HID + n] = (_Float16)(cq[i][r] + vbq);
            kh[(size_t)m * HID + n] = (_Float16)(ck[i][r] + vbk);
            int b = m >> 11, s = m & (SEQ - 1);
            vT[(((size_t)(b * HEADS + h) * HDIM + d) << 11) + s] =
                (_Float16)(cv[i][r] + vbv);
        }
}

// ---------------- kernel 4: simBias = GAMMA*(xn @ xn^T) + maskBig, 64x64/wave ----
__global__ void __launch_bounds__(256, 2)
k_sim(const _Float16* __restrict__ xnh,
      const unsigned char* __restrict__ mask,
      float* __restrict__ simBias) {
    int wave = (blockIdx.x * blockDim.x + threadIdx.x) >> 5;   // 0..2047
    int b   = wave >> 10;
    int rem = wave & 1023;
    int iB = rem >> 5, jB = rem & 31;     // 32x32 blocks of 64x64
    int i0 = iB * 64, j0 = jB * 64;
    const _Float16* xb = xnh + (size_t)b * SEQ * HID;

    v8f acc[4][4];
#pragma unroll
    for (int ci = 0; ci < 4; ++ci)
#pragma unroll
        for (int cj = 0; cj < 4; ++cj) acc[ci][cj] = {};

    for (int kb = 0; kb < HID; kb += 32) {
        v16h a[4], bb[4];
#pragma unroll
        for (int ci = 0; ci < 4; ++ci) a[ci]  = load_a_frag(xb, HID, i0 + ci * 16, kb);
#pragma unroll
        for (int cj = 0; cj < 4; ++cj) bb[cj] = load_b_frag(xb, HID, j0 + cj * 16, kb);
#pragma unroll
        for (int ci = 0; ci < 4; ++ci)
#pragma unroll
            for (int cj = 0; cj < 4; ++cj)
                acc[ci][cj] = wmma_f16(a[ci], bb[cj], acc[ci][cj]);
    }
    int lane = threadIdx.x & 31;
    int colLane = lane & 15, halfSel = lane >> 4;
    float* simb = simBias + (size_t)b * SEQ * SEQ;
#pragma unroll
    for (int ci = 0; ci < 4; ++ci)
#pragma unroll
        for (int cj = 0; cj < 4; ++cj) {
            int j = j0 + cj * 16 + colLane;
#pragma unroll
            for (int r = 0; r < 8; ++r) {
                int i = i0 + ci * 16 + r + 8 * halfSel;
                float bias = GAMMA_C * acc[ci][cj][r];
                if (mask[(size_t)i * SEQ + j]) bias += MASK_BIG;
                simb[(size_t)i * SEQ + j] = bias;
            }
        }
}

// ---------------- kernel 5: flash attention with precomputed bias ----------------
__global__ void __launch_bounds__(256, 2)
k_attn(const _Float16* __restrict__ qh,
       const _Float16* __restrict__ kh,
       const _Float16* __restrict__ vT,
       const float* __restrict__ simBias,
       _Float16* __restrict__ ctxh) {
    __shared__ __align__(32) _Float16 pbuf[8][16 * 32];   // per-wave P staging
    int widx = threadIdx.x >> 5;
    int lane = threadIdx.x & 31;
    int colLane = lane & 15, halfSel = lane >> 4;

    int wave = blockIdx.x * 8 + widx;    // 0..4095
    int qT = wave & 127;
    int h  = (wave >> 7) & (HEADS - 1);
    int b  = wave >> 11;

    int m0 = b * SEQ + qT * 16;                    // absolute row in qh/kh
    const _Float16* qbase = qh + h * HDIM;         // row m -> +m*HID, d contiguous
    const _Float16* kbase = kh + h * HDIM;
    const _Float16* vbase = vT + (size_t)(b * HEADS + h) * HDIM * SEQ; // [d][s]
    const float* simb = simBias + (size_t)b * SEQ * SEQ;

    float mrow[8], lrow[8];
#pragma unroll
    for (int r = 0; r < 8; ++r) { mrow[r] = -1e30f; lrow[r] = 0.f; }
    v8f ctx0 = {}, ctx1 = {}, ctx2 = {}, ctx3 = {};

    // q fragments for K(head-dim)=64 -> two K=32 fragments, loop-invariant
    v16h qa0 = load_a_frag(qbase, HID, m0, 0);
    v16h qa1 = load_a_frag(qbase, HID, m0, 32);

    for (int j0 = 0; j0 < SEQ; j0 += 32) {
        float sc[2][8];
#pragma unroll
        for (int t = 0; t < 2; ++t) {
            int jb = b * SEQ + j0 + t * 16;
            v16h kf0 = load_b_frag(kbase, HID, jb, 0);
            v16h kf1 = load_b_frag(kbase, HID, jb, 32);
            v8f c = {};
            c = wmma_f16(qa0, kf0, c);
            c = wmma_f16(qa1, kf1, c);
            int jcol = j0 + t * 16 + colLane;
#pragma unroll
            for (int r = 0; r < 8; ++r) {
                int qrow = qT * 16 + r + 8 * halfSel;
                sc[t][r] = c[r] * INV_SQRT_HD - simb[(size_t)qrow * SEQ + jcol];
            }
        }
        // online softmax: row max across the 16 lanes of each half-wave
        float scale[8];
#pragma unroll
        for (int r = 0; r < 8; ++r) {
            float v = fmaxf(sc[0][r], sc[1][r]);
            for (int ofs = 1; ofs < 16; ofs <<= 1)
                v = fmaxf(v, __shfl_xor(v, ofs, 32));
            float mnew = fmaxf(mrow[r], v);
            scale[r] = __expf(mrow[r] - mnew);
            mrow[r] = mnew;
        }
#pragma unroll
        for (int t = 0; t < 2; ++t)
#pragma unroll
            for (int r = 0; r < 8; ++r)
                sc[t][r] = __expf(sc[t][r] - mrow[r]);
#pragma unroll
        for (int r = 0; r < 8; ++r) {
            float s = sc[0][r] + sc[1][r];
            for (int ofs = 1; ofs < 16; ofs <<= 1)
                s += __shfl_xor(s, ofs, 32);
            lrow[r] = lrow[r] * scale[r] + s;
            ctx0[r] *= scale[r]; ctx1[r] *= scale[r];
            ctx2[r] *= scale[r]; ctx3[r] *= scale[r];
        }
        // transpose P (C-layout) -> A-fragment via per-wave LDS tile [16][32]
        _Float16* pb = pbuf[widx];
#pragma unroll
        for (int t = 0; t < 2; ++t)
#pragma unroll
            for (int r = 0; r < 8; ++r)
                pb[(r + 8 * halfSel) * 32 + t * 16 + colLane] = (_Float16)sc[t][r];
        __builtin_amdgcn_wave_barrier();
        asm volatile("s_wait_dscnt 0" ::: "memory");
        v16h pa;
        {
            const _Float16* pr = pb + colLane * 32 + halfSel * 8;
            v8h lo = *(const v8h*)pr;
            v8h hi = *(const v8h*)(pr + 16);
#pragma unroll
            for (int i = 0; i < 8; ++i) { pa[i] = lo[i]; pa[i + 8] = hi[i]; }
        }
        __builtin_amdgcn_wave_barrier();
        // ctx(16x64) += P(16x32) @ V(32x64); V B-fragments contiguous from vT
        ctx0 = wmma_f16(pa, load_b_frag(vbase, SEQ,  0, j0), ctx0);
        ctx1 = wmma_f16(pa, load_b_frag(vbase, SEQ, 16, j0), ctx1);
        ctx2 = wmma_f16(pa, load_b_frag(vbase, SEQ, 32, j0), ctx2);
        ctx3 = wmma_f16(pa, load_b_frag(vbase, SEQ, 48, j0), ctx3);
    }
    // finalize: ctx / l -> ctxh[m][h*64 + d]
#pragma unroll
    for (int r = 0; r < 8; ++r) {
        int m = m0 + r + 8 * halfSel;
        float inv = 1.0f / lrow[r];
        _Float16* orow = ctxh + (size_t)m * HID + h * HDIM;
        orow[ 0 + colLane] = (_Float16)(ctx0[r] * inv);
        orow[16 + colLane] = (_Float16)(ctx1[r] * inv);
        orow[32 + colLane] = (_Float16)(ctx2[r] * inv);
        orow[48 + colLane] = (_Float16)(ctx3[r] * inv);
    }
}

// ---------------- kernel 6: output projection, 64x64 per wave (f32 out) --------
__global__ void __launch_bounds__(256, 2)
k_out(const _Float16* __restrict__ ctxh,
      const _Float16* __restrict__ wo,
      const float* __restrict__ bo,
      float* __restrict__ out) {
    int wave = (blockIdx.x * blockDim.x + threadIdx.x) >> 5;   // 0..1023
    int mB = wave >> 4, nB = wave & 15;
    int m0 = mB * 64, n0 = nB * 64;

    v8f acc[4][4];
#pragma unroll
    for (int ci = 0; ci < 4; ++ci)
#pragma unroll
        for (int cj = 0; cj < 4; ++cj) acc[ci][cj] = {};

    for (int kb = 0; kb < HID; kb += 32) {
        v16h a[4], f[4];
#pragma unroll
        for (int ci = 0; ci < 4; ++ci) a[ci] = load_a_frag(ctxh, HID, m0 + ci * 16, kb);
#pragma unroll
        for (int cj = 0; cj < 4; ++cj) f[cj] = load_b_frag(wo, HID, n0 + cj * 16, kb);
#pragma unroll
        for (int ci = 0; ci < 4; ++ci)
#pragma unroll
            for (int cj = 0; cj < 4; ++cj)
                acc[ci][cj] = wmma_f16(a[ci], f[cj], acc[ci][cj]);
    }
    int lane = threadIdx.x & 31;
    int colLane = lane & 15, halfSel = lane >> 4;
#pragma unroll
    for (int cj = 0; cj < 4; ++cj) {
        int n = n0 + cj * 16 + colLane;
        float bias = bo[n];
#pragma unroll
        for (int ci = 0; ci < 4; ++ci)
#pragma unroll
            for (int r = 0; r < 8; ++r) {
                int m = m0 + ci * 16 + r + 8 * halfSel;
                out[(size_t)m * HID + n] = acc[ci][cj][r] + bias;
            }
    }
}

// ---------------- launch ----------------
extern "C" void kernel_launch(void* const* d_in, const int* in_sizes, int n_in,
                              void* d_out, int out_size, void* d_ws, size_t ws_size,
                              hipStream_t stream) {
    const float* x  = (const float*)d_in[0];
    const unsigned char* mask = (const unsigned char*)d_in[1];
    const float* Wq = (const float*)d_in[2];
    const float* bq = (const float*)d_in[3];
    const float* Wk = (const float*)d_in[4];
    const float* bk = (const float*)d_in[5];
    const float* Wv = (const float*)d_in[6];
    const float* bv = (const float*)d_in[7];
    const float* Wo = (const float*)d_in[8];
    const float* bo = (const float*)d_in[9];
    float* out = (float*)d_out;

    // workspace carve-up (256B aligned slices)
    size_t off = 0;
    auto alloc = [&](size_t bytes) {
        void* p = (char*)d_ws + off;
        off += (bytes + 255) & ~(size_t)255;
        return p;
    };
    _Float16* xh   = (_Float16*)alloc((size_t)ROWS * HID * 2);
    _Float16* xnh  = (_Float16*)alloc((size_t)ROWS * HID * 2);
    _Float16* wqh  = (_Float16*)alloc((size_t)HID * HID * 2);
    _Float16* wkh  = (_Float16*)alloc((size_t)HID * HID * 2);
    _Float16* wvh  = (_Float16*)alloc((size_t)HID * HID * 2);
    _Float16* woh  = (_Float16*)alloc((size_t)HID * HID * 2);
    _Float16* qh   = (_Float16*)alloc((size_t)ROWS * HID * 2);
    _Float16* kh   = (_Float16*)alloc((size_t)ROWS * HID * 2);
    _Float16* vT   = (_Float16*)alloc((size_t)ROWS * HID * 2);
    float*    simB = (float*)   alloc((size_t)BATCH * SEQ * SEQ * 4);
    _Float16* ctxh = (_Float16*)alloc((size_t)ROWS * HID * 2);
    (void)ws_size; (void)n_in; (void)in_sizes; (void)out_size;

    dim3 blk(256);

    // 1) convert + normalize x
    k_prep<<<dim3(ROWS), blk, 0, stream>>>(x, xh, xnh);

    // 2) weights -> f16
    const int WN = HID * HID;
    k_cvt<<<dim3(WN / 256), blk, 0, stream>>>(Wq, wqh, WN);
    k_cvt<<<dim3(WN / 256), blk, 0, stream>>>(Wk, wkh, WN);
    k_cvt<<<dim3(WN / 256), blk, 0, stream>>>(Wv, wvh, WN);
    k_cvt<<<dim3(WN / 256), blk, 0, stream>>>(Wo, woh, WN);

    // 3) fused QKV GEMM: 4096 wave-tiles (64x16 x3) / 8 waves per block
    k_qkv<<<dim3(512), blk, 0, stream>>>(xh, wqh, wkh, wvh, bq, bk, bv, qh, kh, vT);

    // 4) cosine-similarity bias (incl. mask fold): 2048 wave-tiles of 64x64
    k_sim<<<dim3(256), blk, 0, stream>>>(xnh, mask, simB);

    // 5) flash attention with diversity bias: 4096 waves
    k_attn<<<dim3(512), blk, 0, stream>>>(qh, kh, vT, simB, ctxh);

    // 6) output projection: 1024 wave-tiles of 64x64
    k_out<<<dim3(128), blk, 0, stream>>>(ctxh, woh, bo, out);
}
